// SGCAM_67233418051955
// MI455X (gfx1250) — matrix operand order repaired and checked by reading, not compiled, not run on hardware
//
#include <hip/hip_runtime.h>
#include <hip/hip_bf16.h>

// ---------------------------------------------------------------------------
// SGCAM non-local block on MI455X (gfx1250, wave32, WMMA + TDM).
// Re-associated:  y = Theta^T (Phi G^T) / N ;  wy = (w_w S^T / N) Theta
// All GEMMs: v_wmma_f32_16x16x32_bf16, fp32 accumulate.
// GEMM operand tiles (all but final-kernel B) are K-contiguous bf16 row blocks staged into padded
// LDS by the Tensor Data Mover (tensor_load_to_lds + s_wait_tensorcnt),
// double-buffered so DMA overlaps WMMA.
// ---------------------------------------------------------------------------

typedef __attribute__((ext_vector_type(16))) __bf16       v16bf;
typedef __attribute__((ext_vector_type(8)))  float        v8f;
typedef __attribute__((ext_vector_type(4)))  unsigned int u32x4;
typedef __attribute__((ext_vector_type(2)))  unsigned int u32x2;
typedef __attribute__((ext_vector_type(4)))  float        f32x4;
typedef __attribute__((ext_vector_type(8)))  int          i32x8;
typedef __attribute__((ext_vector_type(4)))  int          i32x4;

static constexpr int kB  = 16;
static constexpr int kC  = 1024;
static constexpr int kCI = 512;
static constexpr int kN  = 2048;
static constexpr float kEPS = 1e-5f;

static constexpr int LDT = 40;   // LDS tile row stride in ushorts (64B data + 16B pad)

#if defined(__AMDGCN__) && __has_builtin(__builtin_amdgcn_tensor_load_to_lds)
  #define HAVE_TDM 1
#else
  #define HAVE_TDM 0
#endif
#if HAVE_TDM && __has_include(<hip/amd_detail/amd_gfx1250_TDM.h>)
  #define TDM_6ARG 1
#else
  #define TDM_6ARG 0
#endif

__device__ __forceinline__ unsigned short f32_to_bf16(float f) {
    union { float f; unsigned int u; } c; c.f = f;
    unsigned int u = c.u;
    u += 0x7FFFu + ((u >> 16) & 1u);     // round-to-nearest-even
    return (unsigned short)(u >> 16);
}

union FragU { u32x4 q[2]; v16bf v; };

// ---------------- tile staging: global (bf16, K-contiguous rows) -> LDS ---------
// Manual path: 256 threads copy a 128x32-bf16 tile into padded LDS rows.
__device__ __forceinline__ void ld_bf16_direct(unsigned short* dst, const unsigned short* src,
                                               int srcLd, int tid) {
    int seg  = tid & 7;          // 8 segments * 4 bf16 = 32 K
    int row0 = tid >> 3;         // 32 rows / pass
#pragma unroll
    for (int p = 0; p < 4; ++p) {
        int row = row0 + p * 32;
        u32x2 u = *(const u32x2*)(src + (size_t)row * srcLd + seg * 4);
        *(u32x2*)(dst + row * LDT + seg * 4) = u;
    }
}
// bf16 source laid out [32 K][128 cols] -> transposed store (final kernel B only)
__device__ __forceinline__ void ld_bf16_trans(unsigned short* dst, const unsigned short* src,
                                              int srcLd, int tid) {
    int n0 = (tid & 31) * 4;
    int k0 = tid >> 5;
#pragma unroll
    for (int p = 0; p < 4; ++p) {
        int k = k0 + p * 8;
        u32x2 u = *(const u32x2*)(src + (size_t)k * srcLd + n0);
        dst[(n0 + 0) * LDT + k] = (unsigned short)(u.x & 0xFFFFu);
        dst[(n0 + 1) * LDT + k] = (unsigned short)(u.x >> 16);
        dst[(n0 + 2) * LDT + k] = (unsigned short)(u.y & 0xFFFFu);
        dst[(n0 + 3) * LDT + k] = (unsigned short)(u.y >> 16);
    }
}

#if HAVE_TDM
// TDM: 2D descriptor, tile 32 (K, dim0) x 128 (rows, dim1), 2-byte elements,
// LDS padding: after every 16 DWORDs (64B row) insert 4 DWORDs (16B) -> LDT=40.
__device__ __forceinline__ void tdm_load_tile(const unsigned short* src, int srcLd,
                                              unsigned short* lds) {
    unsigned long long ga = (unsigned long long)(const void*)src;
    unsigned lds_off = (unsigned)(unsigned long long)(void*)lds;  // low 32 bits = LDS offset
    u32x4 g0;
    g0.x = 0x1u;                                                  // count=1, user descriptor
    g0.y = lds_off;                                               // lds_addr
    g0.z = (unsigned)(ga & 0xFFFFFFFFu);                          // global_addr[31:0]
    g0.w = (unsigned)((ga >> 32) & 0x01FFFFFFu) | (2u << 30);     // ga[56:32] | type=2
    i32x8 g1;
    g1[0] = (int)((1u << 16) | (1u << 20) | (3u << 22) | (3u << 25));
            // data_size=1(2B) | pad_enable | pad_interval=3(16DW) | pad_amount=3(4DW)
    g1[1] = (int)(((unsigned)srcLd & 0xFFFFu) << 16);             // tensor_dim0[15:0]
    g1[2] = (int)((((unsigned)srcLd >> 16) & 0xFFFFu) | (0xFFFFu << 16));
            // tensor_dim0[31:16] | tensor_dim1[15:0] = 65535 (no OOB)
    g1[3] = (int)(32u << 16);                                     // tensor_dim1 hi=0 | tile_dim0=32
    g1[4] = (int)128u;                                            // tile_dim1=128 (tile_dim2=0)
    g1[5] = (int)srcLd;                                           // tensor_dim0_stride[31:0]
    g1[6] = 0;                                                    // stride hi | dim1_stride lo
    g1[7] = 0;
    i32x4 z4 = (i32x4)0;
#if TDM_6ARG
    i32x8 z8 = (i32x8)0;
    __builtin_amdgcn_tensor_load_to_lds(g0, g1, z4, z4, z8, 0);
#else
    __builtin_amdgcn_tensor_load_to_lds(g0, g1, z4, z4, 0);
#endif
}
#endif

// Stage one 128x32 tile; TDM issued by wave 0 only (EXEC-independent, per-wave op).
__device__ __forceinline__ void stage_direct(unsigned short* lds, const unsigned short* src,
                                             int srcLd, int tid) {
#if HAVE_TDM
    if ((tid >> 5) == 0) tdm_load_tile(src, srcLd, lds);
#else
    ld_bf16_direct(lds, src, srcLd, tid);
#endif
}
__device__ __forceinline__ void stage_sync(int wid) {
#if HAVE_TDM && __has_builtin(__builtin_amdgcn_s_wait_tensorcnt)
    if (wid == 0) __builtin_amdgcn_s_wait_tensorcnt(0);
#endif
    __syncthreads();
}

// ---------------- LDS -> VGPR fragments: A and B both use k = 16*(lane/16) + s ------------
__device__ __forceinline__ v16bf frag_a(const unsigned short* As, int mBase, int lane) {
    int m  = mBase + (lane & 15);
    int hi = (lane >> 4) & 1;
    FragU f;
    f.q[0] = *(const u32x4*)(As + m * LDT + 16 * hi);
    f.q[1] = *(const u32x4*)(As + m * LDT + 16 * hi + 8);
    return f.v;
}
__device__ __forceinline__ v16bf frag_b(const unsigned short* Bs, int nBase, int lane) {
    int n  = nBase + (lane & 15);
    int hi = (lane >> 4) & 1;
    FragU f;
    const u32x4* p = (const u32x4*)(Bs + n * LDT + 16 * hi);
    f.q[0] = p[0];
    f.q[1] = p[1];
    return f.v;
}

// Each wave owns a 64(M) x 32(N) C tile: 4x2 accumulators.
__device__ __forceinline__ void wave_mma(const unsigned short* As, const unsigned short* Bs,
                                         int wm, int wn, int lane, v8f acc[4][2]) {
    v16bf a[4], b[2];
#pragma unroll
    for (int i = 0; i < 4; ++i) a[i] = frag_a(As, wm * 64 + i * 16, lane);
#pragma unroll
    for (int j = 0; j < 2; ++j) b[j] = frag_b(Bs, wn * 32 + j * 16, lane);
#pragma unroll
    for (int i = 0; i < 4; ++i)
#pragma unroll
        for (int j = 0; j < 2; ++j)
            acc[i][j] = __builtin_amdgcn_wmma_f32_16x16x32_bf16(
                false, a[i], false, b[j], (short)0, acc[i][j], false, false);
}

// ---------------------------------------------------------------------------
// Prep 1: fp32 -> bf16 weight conversion (theta_w, phi_w, g_w, w_w; 512K elems each)
// ---------------------------------------------------------------------------
__global__ __launch_bounds__(256) void convW_kernel(
    const float* __restrict__ tw, const float* __restrict__ pw,
    const float* __restrict__ gw, const float* __restrict__ ww,
    unsigned short* __restrict__ otw, unsigned short* __restrict__ opw,
    unsigned short* __restrict__ ogw, unsigned short* __restrict__ oww)
{
    int z = blockIdx.y;
    const float* s; unsigned short* d;
    if (z == 0)      { s = tw; d = otw; }
    else if (z == 1) { s = pw; d = opw; }
    else if (z == 2) { s = gw; d = ogw; }
    else             { s = ww; d = oww; }
    size_t i = ((size_t)blockIdx.x * 256 + threadIdx.x) * 4;
    f32x4 v = *(const f32x4*)(s + i);
    u32x2 u;
    u.x = (unsigned)f32_to_bf16(v.x) | ((unsigned)f32_to_bf16(v.y) << 16);
    u.y = (unsigned)f32_to_bf16(v.z) | ((unsigned)f32_to_bf16(v.w) << 16);
    *(u32x2*)(d + i) = u;
}

// ---------------------------------------------------------------------------
// Prep 2: transpose + convert  [B,C,N] f32 -> [B,N,C] bf16  (32x32 LDS tiles)
// ---------------------------------------------------------------------------
__global__ __launch_bounds__(256) void convT_kernel(
    const float* __restrict__ src, unsigned short* __restrict__ dst)
{
    __shared__ unsigned short t[32][33];
    int b = blockIdx.z;
    int n0 = blockIdx.x * 32, c0 = blockIdx.y * 32;
    int tid = threadIdx.x;
    int r = tid >> 3, seg = (tid & 7) * 4;
    f32x4 v = *(const f32x4*)(src + ((size_t)b * kC + c0 + r) * kN + n0 + seg);
    t[r][seg + 0] = f32_to_bf16(v.x);
    t[r][seg + 1] = f32_to_bf16(v.y);
    t[r][seg + 2] = f32_to_bf16(v.z);
    t[r][seg + 3] = f32_to_bf16(v.w);
    __syncthreads();
    u32x2 u;
    u.x = (unsigned)t[seg + 0][r] | ((unsigned)t[seg + 1][r] << 16);
    u.y = (unsigned)t[seg + 2][r] | ((unsigned)t[seg + 3][r] << 16);
    *(u32x2*)(dst + ((size_t)b * kN + n0 + r) * kC + c0 + seg) = u;
}

// Per-channel BN folding: alpha = gamma*rsqrt(var+eps), cfix = beta + alpha*(w_b - mean)
__global__ __launch_bounds__(256) void bn_prep_kernel(
    const float* __restrict__ wb, const float* __restrict__ gamma,
    const float* __restrict__ beta, const float* __restrict__ mean,
    const float* __restrict__ var, float* __restrict__ alpha, float* __restrict__ cfix)
{
    int c = blockIdx.x * blockDim.x + threadIdx.x;
    if (c < kC) {
        float a = gamma[c] * rsqrtf(var[c] + kEPS);
        alpha[c] = a;
        cfix[c]  = beta[c] + a * (wb[c] - mean[c]);
    }
}

// ---------------------------------------------------------------------------
// Kernel 1: projections.  Out[ci,n] = W[ci,:]·X[:,n] + bias[ci]  (bf16 out)
// A = Wbf [CI,C], B^T rows = X^T [N,C] -> both direct TDM tiles.
// ---------------------------------------------------------------------------
__global__ __launch_bounds__(256) void proj_kernel(
    const unsigned short* __restrict__ xT, const unsigned short* __restrict__ x2T,
    const unsigned short* __restrict__ twb, const float* __restrict__ tb,
    const unsigned short* __restrict__ pwb, const float* __restrict__ pb,
    const unsigned short* __restrict__ gwb, const float* __restrict__ gb,
    unsigned short* __restrict__ theta, unsigned short* __restrict__ phi,
    unsigned short* __restrict__ g)
{
    __shared__ unsigned short As[2][128 * LDT];
    __shared__ unsigned short Bs[2][128 * LDT];
    int tid = threadIdx.x, lane = tid & 31, wid = tid >> 5;
    int wm = wid >> 2, wn = wid & 3;
    int which = blockIdx.z % 3, b = blockIdx.z / 3;
    const unsigned short* W; const float* bias; const unsigned short* XT; unsigned short* O;
    if (which == 0)      { W = twb; bias = tb; XT = xT;  O = theta; }
    else if (which == 1) { W = pwb; bias = pb; XT = x2T; O = phi; }
    else                 { W = gwb; bias = gb; XT = x2T; O = g; }
    int bm = blockIdx.y * 128, bn = blockIdx.x * 128;
    const unsigned short* Asrc = W + (size_t)bm * kC;
    const unsigned short* Bsrc = XT + ((size_t)b * kN + bn) * kC;
    unsigned short* Ob = O + (size_t)b * kCI * kN;

    v8f acc[4][2] = {};
    stage_direct(As[0], Asrc, kC, tid);
    stage_direct(Bs[0], Bsrc, kC, tid);
    stage_sync(wid);
    const int ktiles = kC / 32;
    for (int kt = 0; kt < ktiles; ++kt) {
        int cur = kt & 1;
        if (kt + 1 < ktiles) {
            stage_direct(As[cur ^ 1], Asrc + (kt + 1) * 32, kC, tid);
            stage_direct(Bs[cur ^ 1], Bsrc + (kt + 1) * 32, kC, tid);
        }
        wave_mma(As[cur], Bs[cur], wm, wn, lane, acc);
        stage_sync(wid);
    }
    int hi = lane >> 4, ln = lane & 15;
#pragma unroll
    for (int i = 0; i < 4; ++i)
#pragma unroll
        for (int j = 0; j < 2; ++j) {
            int gm0 = bm + wm * 64 + i * 16;
            int gn  = bn + wn * 32 + j * 16 + ln;
#pragma unroll
            for (int r = 0; r < 8; ++r) {
                int row = gm0 + r + 8 * hi;
                Ob[(size_t)row * kN + gn] = f32_to_bf16(acc[i][j][r] + bias[row]);
            }
        }
}

// ---------------------------------------------------------------------------
// Kernel 2: S_b[ci1,ci2] = sum_n Phi[ci1,n]*G[ci2,n]  (bf16 out, [CI,CI])
// ---------------------------------------------------------------------------
__global__ __launch_bounds__(256) void s_kernel(
    const unsigned short* __restrict__ phi, const unsigned short* __restrict__ g,
    unsigned short* __restrict__ S)
{
    __shared__ unsigned short As[2][128 * LDT];
    __shared__ unsigned short Bs[2][128 * LDT];
    int tid = threadIdx.x, lane = tid & 31, wid = tid >> 5;
    int wm = wid >> 2, wn = wid & 3;
    int b = blockIdx.z;
    int bm = blockIdx.y * 128, bn = blockIdx.x * 128;
    const unsigned short* Asrc = phi + (size_t)b * kCI * kN + (size_t)bm * kN;
    const unsigned short* Bsrc = g   + (size_t)b * kCI * kN + (size_t)bn * kN;
    unsigned short* Sb = S + (size_t)b * kCI * kCI;

    v8f acc[4][2] = {};
    stage_direct(As[0], Asrc, kN, tid);
    stage_direct(Bs[0], Bsrc, kN, tid);
    stage_sync(wid);
    const int ktiles = kN / 32;
    for (int kt = 0; kt < ktiles; ++kt) {
        int cur = kt & 1;
        if (kt + 1 < ktiles) {
            stage_direct(As[cur ^ 1], Asrc + (kt + 1) * 32, kN, tid);
            stage_direct(Bs[cur ^ 1], Bsrc + (kt + 1) * 32, kN, tid);
        }
        wave_mma(As[cur], Bs[cur], wm, wn, lane, acc);
        stage_sync(wid);
    }
    int hi = lane >> 4, ln = lane & 15;
#pragma unroll
    for (int i = 0; i < 4; ++i)
#pragma unroll
        for (int j = 0; j < 2; ++j) {
            int gm0 = bm + wm * 64 + i * 16;
            int gn  = bn + wn * 32 + j * 16 + ln;
#pragma unroll
            for (int r = 0; r < 8; ++r)
                Sb[(size_t)(gm0 + r + 8 * hi) * kCI + gn] = f32_to_bf16(acc[i][j][r]);
        }
}

// ---------------------------------------------------------------------------
// Kernel 3: M_b[o,c] = (1/N) * sum_k w_w[o,k]*S_b[c,k]  (bf16 out, [C,CI])
// ---------------------------------------------------------------------------
__global__ __launch_bounds__(256) void m_kernel(
    const unsigned short* __restrict__ wwb, const unsigned short* __restrict__ S,
    unsigned short* __restrict__ M)
{
    __shared__ unsigned short As[2][128 * LDT];
    __shared__ unsigned short Bs[2][128 * LDT];
    int tid = threadIdx.x, lane = tid & 31, wid = tid >> 5;
    int wm = wid >> 2, wn = wid & 3;
    int b = blockIdx.z;
    int bm = blockIdx.y * 128, bn = blockIdx.x * 128;
    const unsigned short* Asrc = wwb + (size_t)bm * kCI;
    const unsigned short* Bsrc = S + (size_t)b * kCI * kCI + (size_t)bn * kCI;
    unsigned short* Mb = M + (size_t)b * kC * kCI;

    v8f acc[4][2] = {};
    stage_direct(As[0], Asrc, kCI, tid);
    stage_direct(Bs[0], Bsrc, kCI, tid);
    stage_sync(wid);
    const int ktiles = kCI / 32;
    for (int kt = 0; kt < ktiles; ++kt) {
        int cur = kt & 1;
        if (kt + 1 < ktiles) {
            stage_direct(As[cur ^ 1], Asrc + (kt + 1) * 32, kCI, tid);
            stage_direct(Bs[cur ^ 1], Bsrc + (kt + 1) * 32, kCI, tid);
        }
        wave_mma(As[cur], Bs[cur], wm, wn, lane, acc);
        stage_sync(wid);
    }
    const float scale = 1.0f / (float)kN;
    int hi = lane >> 4, ln = lane & 15;
#pragma unroll
    for (int i = 0; i < 4; ++i)
#pragma unroll
        for (int j = 0; j < 2; ++j) {
            int gm0 = bm + wm * 64 + i * 16;
            int gn  = bn + wn * 32 + j * 16 + ln;
#pragma unroll
            for (int r = 0; r < 8; ++r)
                Mb[(size_t)(gm0 + r + 8 * hi) * kCI + gn] =
                    f32_to_bf16(acc[i][j][r] * scale);
        }
}

// ---------------------------------------------------------------------------
// Kernel 4: out[b,c,n] = alpha[c]*(M_b @ Theta_b)[c,n] + cfix[c] + x[b,c,n]
// A = M (direct TDM); B = Theta [CI,N] staged with LDS transpose.
// ---------------------------------------------------------------------------
__global__ __launch_bounds__(256) void final_kernel(
    const unsigned short* __restrict__ M, const unsigned short* __restrict__ theta,
    const float* __restrict__ alpha, const float* __restrict__ cfix,
    const float* __restrict__ x, float* __restrict__ out)
{
    __shared__ unsigned short As[2][128 * LDT];
    __shared__ unsigned short Bs[2][128 * LDT];
    int tid = threadIdx.x, lane = tid & 31, wid = tid >> 5;
    int wm = wid >> 2, wn = wid & 3;
    int b = blockIdx.z;
    int bm = blockIdx.y * 128, bn = blockIdx.x * 128;
    const unsigned short* Asrc = M + (size_t)b * kC * kCI + (size_t)bm * kCI;
    const unsigned short* Thb  = theta + (size_t)b * kCI * kN;

    v8f acc[4][2] = {};
    stage_direct(As[0], Asrc, kCI, tid);
    ld_bf16_trans(Bs[0], Thb + bn, kN, tid);
    stage_sync(wid);
    const int ktiles = kCI / 32;
    for (int kt = 0; kt < ktiles; ++kt) {
        int cur = kt & 1;
        if (kt + 1 < ktiles) {
            stage_direct(As[cur ^ 1], Asrc + (kt + 1) * 32, kCI, tid);
            ld_bf16_trans(Bs[cur ^ 1], Thb + (size_t)((kt + 1) * 32) * kN + bn, kN, tid);
        }
        wave_mma(As[cur], Bs[cur], wm, wn, lane, acc);
        stage_sync(wid);
    }
    int hi = lane >> 4, ln = lane & 15;
#pragma unroll
    for (int i = 0; i < 4; ++i)
#pragma unroll
        for (int j = 0; j < 2; ++j) {
            int gm0 = bm + wm * 64 + i * 16;
            int gn  = bn + wn * 32 + j * 16 + ln;
#pragma unroll
            for (int r = 0; r < 8; ++r) {
                int row = gm0 + r + 8 * hi;
                size_t idx = ((size_t)b * kC + row) * kN + gn;
                out[idx] = alpha[row] * acc[i][j][r] + cfix[row] + x[idx];
            }
        }
}

// ---------------------------------------------------------------------------
extern "C" void kernel_launch(void* const* d_in, const int* in_sizes, int n_in,
                              void* d_out, int out_size, void* d_ws, size_t ws_size,
                              hipStream_t stream) {
    const float* x   = (const float*)d_in[0];
    const float* x2  = (const float*)d_in[1];
    const float* tw  = (const float*)d_in[2];
    const float* tb  = (const float*)d_in[3];
    const float* pw  = (const float*)d_in[4];
    const float* pb  = (const float*)d_in[5];
    const float* gw  = (const float*)d_in[6];
    const float* gb  = (const float*)d_in[7];
    const float* ww  = (const float*)d_in[8];
    const float* wb  = (const float*)d_in[9];
    const float* bng = (const float*)d_in[10];
    const float* bnb = (const float*)d_in[11];
    const float* bnm = (const float*)d_in[12];
    const float* bnv = (const float*)d_in[13];
    float* out = (float*)d_out;

    // workspace layout (all bf16 except BN scalars)
    unsigned short* xT    = (unsigned short*)d_ws;                 // [B,N,C]
    unsigned short* x2T   = xT  + (size_t)kB * kN * kC;            // [B,N,C]
    unsigned short* twb   = x2T + (size_t)kB * kN * kC;            // [CI,C]
    unsigned short* pwb   = twb + (size_t)kCI * kC;
    unsigned short* gwb   = pwb + (size_t)kCI * kC;
    unsigned short* wwb   = gwb + (size_t)kCI * kC;                // [C,CI]
    unsigned short* theta = wwb + (size_t)kC * kCI;                // [B,CI,N]
    unsigned short* phi   = theta + (size_t)kB * kCI * kN;
    unsigned short* g     = phi   + (size_t)kB * kCI * kN;
    unsigned short* S     = g     + (size_t)kB * kCI * kN;         // [B,CI,CI]
    unsigned short* Mm    = S     + (size_t)kB * kCI * kCI;        // [B,C,CI]
    float*          alpha = (float*)(Mm + (size_t)kB * kC * kCI);  // [C]
    float*          cfix  = alpha + kC;                            // [C]

    convW_kernel<<<dim3((kCI * kC) / (256 * 4), 4), dim3(256), 0, stream>>>(
        tw, pw, gw, ww, twb, pwb, gwb, wwb);
    convT_kernel<<<dim3(kN / 32, kC / 32, kB), dim3(256), 0, stream>>>(x,  xT);
    convT_kernel<<<dim3(kN / 32, kC / 32, kB), dim3(256), 0, stream>>>(x2, x2T);
    bn_prep_kernel<<<dim3((kC + 255) / 256), dim3(256), 0, stream>>>(
        wb, bng, bnb, bnm, bnv, alpha, cfix);

    proj_kernel<<<dim3(kN / 128, kCI / 128, 3 * kB), dim3(256), 0, stream>>>(
        xT, x2T, twb, tb, pwb, pb, gwb, gb, theta, phi, g);

    s_kernel<<<dim3(kCI / 128, kCI / 128, kB), dim3(256), 0, stream>>>(phi, g, S);

    m_kernel<<<dim3(kCI / 128, kC / 128, kB), dim3(256), 0, stream>>>(wwb, S, Mm);

    final_kernel<<<dim3(kN / 128, kC / 128, kB), dim3(256), 0, stream>>>(
        Mm, theta, alpha, cfix, x, out);
}
